// Graph_Representation_Learning_68436008894714
// MI455X (gfx1250) — matrix operand order repaired and checked by reading, not compiled, not run on hardware
//
#include <hip/hip_runtime.h>
#include <hip/hip_bf16.h>

// ---------------------------------------------------------------------------
// GIN graph representation learning on gfx1250 (MI455X).
// GEMMs run through v_wmma_f32_16x16x32_bf16 (bf16 in, f32 accumulate).
// K==128 GEMMs: whole 128x128 A and B tiles staged in 64KB LDS (XOR-swizzled,
// bank-conflict-free) -> barrier-free inner loop of 32 WMMAs.
// K==512 GEMMs: 32-wide k-slabs with B-fragment preload + global prefetch.
// Edge scatter / pooling use global_atomic_add_f32 (bandwidth-bound part).
// ---------------------------------------------------------------------------

#define NNODES  100000
#define NEDGES  600000
#define DIMC    128
#define NGRAPHS 1000
#define EDIM    512   // DIM * N_LAYERS

typedef __attribute__((ext_vector_type(16))) __bf16 v16bf;
typedef __attribute__((ext_vector_type(8)))  float  v8f;

__device__ __forceinline__ unsigned short f2bf(float f) {
  unsigned int u = __float_as_uint(f);
  u += 0x7fffu + ((u >> 16) & 1u);          // round-to-nearest-even
  return (unsigned short)(u >> 16);
}
__device__ __forceinline__ unsigned int pack2(float a, float b) {
  return (unsigned int)f2bf(a) | ((unsigned int)f2bf(b) << 16);
}

// XOR swizzle on 16-byte granules: row-major [row][k] (k in shorts, 0..127),
// granule index (k>>3) xored with (row & 15) -> conflict-free ds_load_b128.
__device__ __forceinline__ int swz(int row, int k) {
  return row * 128 + ((((k >> 3) ^ (row & 15)) << 3) | (k & 7));
}

union FragBF { uint4 q[2]; v16bf v; };

// ---------------------------------------------------------------------------
// Generic GEMM: out = act( (A [+A2]) @ B + bias [+ Cadd] )
// A: [M,K] f32 row-major, B: [K,N] f32 row-major, out: [M,N] f32.
// Block tile 128x128; 8 waves, each wave one 16(M) x 128(N) strip.
// ---------------------------------------------------------------------------
__global__ __launch_bounds__(256)
void gemm_wmma_bf16(const float* __restrict__ A, const float* __restrict__ A2,
                    const float* __restrict__ B, const float* __restrict__ bias,
                    const float* __restrict__ Cadd, float* __restrict__ out,
                    int M, int N, int K, int doLrelu)
{
  __shared__ unsigned short sMem[32768];    // 64 KB, aliased by both paths

  const int tid   = threadIdx.x;
  const int lane  = tid & 31;
  const int wid   = tid >> 5;
  const int lm    = lane & 15;
  const int lh    = lane >> 4;              // half-wave: 0 or 1
  const int mrow0 = blockIdx.x * 128;
  const int nb0   = blockIdx.y * 128;

  v8f acc[8] = {};

  if (K == 128) {
    // =================== fully-staged path (node GEMMs) ====================
    unsigned short* sA = sMem;              // [128 rows][128 k], swizzled
    unsigned short* sB = sMem + 16384;      // [128 n  ][128 k], swizzled

    // stage A (+A2): 128 rows x 128 k, f32 -> bf16
    #pragma unroll
    for (int j = 0; j < 16; ++j) {
      int i   = tid + 256 * j;              // 0..4095 over 128 rows x 32 float4
      int row = i >> 5;
      int c4  = (i & 31) << 2;
      int gr  = mrow0 + row; if (gr >= M) gr = M - 1;
      float4 v = *(const float4*)(A + (size_t)gr * 128 + c4);
      if (A2) {
        const float4 b = *(const float4*)(A2 + (size_t)gr * 128 + c4);
        v.x += b.x; v.y += b.y; v.z += b.z; v.w += b.w;
      }
      *(uint2*)&sA[swz(row, c4)] = make_uint2(pack2(v.x, v.y), pack2(v.z, v.w));
    }
    // stage B transposed: [n][k], f32 -> bf16
    #pragma unroll
    for (int j = 0; j < 16; ++j) {
      int i  = tid + 256 * j;               // 0..4095 over 128 k x 32 n-groups
      int k  = i >> 5;
      int n4 = (i & 31) << 2;
      const float4 b = *(const float4*)(B + (size_t)k * N + nb0 + n4);
      sB[swz(n4 + 0, k)] = f2bf(b.x);
      sB[swz(n4 + 1, k)] = f2bf(b.y);
      sB[swz(n4 + 2, k)] = f2bf(b.z);
      sB[swz(n4 + 3, k)] = f2bf(b.w);
    }
    __syncthreads();                        // the only barrier

    const int arow = wid * 16 + lm;
    const int ar15 = arow & 15;
    #pragma unroll
    for (int k0 = 0; k0 < 128; k0 += 32) {
      // A fragment: lanes 0-15 K {k0..k0+7, k0+16..23}; lanes 16-31 shifted by 8
      FragBF fa;
      {
        int g0 = (k0 >> 3) + lh;            // granule of first 8 K values
        fa.q[0] = *(const uint4*)&sA[arow * 128 + (((g0    ) ^ ar15) << 3)];
        fa.q[1] = *(const uint4*)&sA[arow * 128 + (((g0 + 2) ^ ar15) << 3)];
      }
      // preload all 8 B fragments so WMMAs don't each drain DScnt
      FragBF fb[8];
      #pragma unroll
      for (int nt = 0; nt < 8; ++nt) {
        int br  = nt * 16 + lm;
        int br15 = br & 15;
        int g0  = (k0 >> 3) + lh;           // lanes 0-15: K {k0..k0+7, k0+16..23}; 16-31: +8
        fb[nt].q[0] = *(const uint4*)&sB[br * 128 + (((g0    ) ^ br15) << 3)];
        fb[nt].q[1] = *(const uint4*)&sB[br * 128 + (((g0 + 2) ^ br15) << 3)];
      }
      #pragma unroll
      for (int nt = 0; nt < 8; ++nt)
        acc[nt] = __builtin_amdgcn_wmma_f32_16x16x32_bf16(
            false, fa.v, false, fb[nt].v, (short)0, acc[nt], false, false);
    }
  } else {
    // =================== k-slab path (FF GEMMs, K=512) =====================
    unsigned short* sA = sMem;              // [128 rows][40]
    unsigned short* sB = sMem + 128 * 40;   // [128 n  ][40]

    for (int k0 = 0; k0 < K; k0 += 32) {
      #pragma unroll
      for (int j = 0; j < 4; ++j) {
        int i   = tid + 256 * j;
        int row = i >> 3;
        int c4  = (i & 7) << 2;
        int gr  = mrow0 + row; if (gr >= M) gr = M - 1;
        float4 v = *(const float4*)(A + (size_t)gr * K + k0 + c4);
        if (A2) {
          const float4 b = *(const float4*)(A2 + (size_t)gr * K + k0 + c4);
          v.x += b.x; v.y += b.y; v.z += b.z; v.w += b.w;
        }
        if (k0 + 32 < K)                    // prefetch next slab (global_prefetch_b8)
          __builtin_prefetch(A + (size_t)gr * K + k0 + 32 + c4, 0, 3);
        *(uint2*)&sA[row * 40 + c4] = make_uint2(pack2(v.x, v.y), pack2(v.z, v.w));
      }
      #pragma unroll
      for (int j = 0; j < 4; ++j) {
        int i  = tid + 256 * j;
        int kl = i >> 5;
        int n4 = (i & 31) << 2;
        const float4 b = *(const float4*)(B + (size_t)(k0 + kl) * N + nb0 + n4);
        if (k0 + 32 < K)
          __builtin_prefetch(B + (size_t)(k0 + 32 + kl) * N + nb0 + n4, 0, 3);
        sB[(n4 + 0) * 40 + kl] = f2bf(b.x);
        sB[(n4 + 1) * 40 + kl] = f2bf(b.y);
        sB[(n4 + 2) * 40 + kl] = f2bf(b.z);
        sB[(n4 + 3) * 40 + kl] = f2bf(b.w);
      }
      __syncthreads();

      FragBF fa;
      {
        const unsigned short* pa = &sA[(wid * 16 + lm) * 40 + lh * 8];
        fa.q[0] = *(const uint4*)pa;
        fa.q[1] = *(const uint4*)(pa + 16);
      }
      FragBF fb[8];
      #pragma unroll
      for (int nt = 0; nt < 8; ++nt) {
        const unsigned short* pb = &sB[(nt * 16 + lm) * 40 + lh * 8];
        fb[nt].q[0] = *(const uint4*)pb;
        fb[nt].q[1] = *(const uint4*)(pb + 16);
      }
      #pragma unroll
      for (int nt = 0; nt < 8; ++nt)
        acc[nt] = __builtin_amdgcn_wmma_f32_16x16x32_bf16(
            false, fa.v, false, fb[nt].v, (short)0, acc[nt], false, false);
      __syncthreads();
    }
  }

  // ---- epilogue: C/D layout: VGPR r -> row r (lanes 0-15) / r+8 (lanes 16-31)
  const int rbase = mrow0 + wid * 16 + (lh ? 8 : 0);
  #pragma unroll
  for (int nt = 0; nt < 8; ++nt) {
    int col  = nb0 + nt * 16 + lm;
    float bv = bias ? bias[col] : 0.f;
    #pragma unroll
    for (int r = 0; r < 8; ++r) {
      int row = rbase + r;
      if (row < M) {
        float v = acc[nt][r] + bv;
        if (Cadd) v += Cadd[(size_t)row * N + col];
        if (doLrelu) v = v > 0.f ? v : 0.01f * v;
        out[(size_t)row * N + col] = v;
      }
    }
  }
}

// ---------------------------------------------------------------------------
// agg[dst[e], :] += h[src[e], :]    (two edges per 256-thread block)
// ---------------------------------------------------------------------------
__global__ __launch_bounds__(256)
void scatter_add(const float* __restrict__ h, const long long* __restrict__ ei,
                 float* __restrict__ agg, int nE)
{
  int e = blockIdx.x * 2 + (threadIdx.x >> 7);
  int c = threadIdx.x & 127;
  if (e < nE) {
    int s = (int)ei[e];
    int d = (int)ei[(size_t)nE + e];
    atomicAdd(&agg[(size_t)d * DIMC + c], h[(size_t)s * DIMC + c]);
  }
}

__global__ void zero_f32(float* __restrict__ p, int n)
{
  int i = blockIdx.x * blockDim.x + threadIdx.x;
  if (i < n) p[i] = 0.f;
}

// per-column sum / sum-of-squares into stats[0..127] / stats[128..255]
__global__ __launch_bounds__(256)
void col_stats(const float* __restrict__ h2, float* __restrict__ stats, int nrows)
{
  int c = threadIdx.x & 127;
  int r = blockIdx.x * 2 + (threadIdx.x >> 7);
  float s = 0.f, sq = 0.f;
  for (; r < nrows; r += gridDim.x * 2) {
    float v = h2[(size_t)r * DIMC + c];
    s += v; sq += v * v;
  }
  atomicAdd(&stats[c], s);
  atomicAdd(&stats[DIMC + c], sq);
}

// fold BatchNorm (training, biased var, eps=1e-4) into scale/shift per column
__global__ void bn_finalize(const float* __restrict__ stats, const float* __restrict__ g,
                            const float* __restrict__ b, float* __restrict__ ss, int nrows)
{
  int c = threadIdx.x;
  float inv_n = 1.f / (float)nrows;
  float mean  = stats[c] * inv_n;
  float var   = stats[DIMC + c] * inv_n - mean * mean;
  float sc    = g[c] * rsqrtf(var + 1e-4f);
  ss[c]        = sc;
  ss[DIMC + c] = b[c] - mean * sc;
}

// h = h2*scale+shift ; fused global_add_pool: y[batch[r], layer*128 + c] += h
__global__ __launch_bounds__(256)
void bn_apply_pool(const float* __restrict__ h2, const float* __restrict__ ss,
                   const long long* __restrict__ batch, float* __restrict__ h,
                   float* __restrict__ y, int layer, int nrows)
{
  int i = blockIdx.x * 256 + threadIdx.x;
  if (i >= nrows * DIMC) return;
  int r = i >> 7, c = i & 127;
  float v = h2[i] * ss[c] + ss[DIMC + c];
  h[i] = v;
  int g = (int)batch[r];
  atomicAdd(&y[(size_t)g * EDIM + layer * DIMC + c], v);
}

// ---------------------------------------------------------------------------
extern "C" void kernel_launch(void* const* d_in, const int* in_sizes, int n_in,
                              void* d_out, int out_size, void* d_ws, size_t ws_size,
                              hipStream_t stream)
{
  (void)in_sizes; (void)n_in; (void)out_size; (void)ws_size;

  const float*     x       = (const float*)d_in[0];
  const long long* ei      = (const long long*)d_in[1];   // int64 [2, E]
  const long long* bat     = (const long long*)d_in[2];   // int64 [N]
  const float*     pre_w   = (const float*)d_in[3];
  const float*     pre_b   = (const float*)d_in[4];
  const float*     conv_w1 = (const float*)d_in[5];       // [4,128,128]
  const float*     conv_w2 = (const float*)d_in[6];
  const float*     bn_g    = (const float*)d_in[7];       // [4,128]
  const float*     bn_b    = (const float*)d_in[8];
  const float*     ff_w1   = (const float*)d_in[9];
  const float*     ff_w2   = (const float*)d_in[10];
  const float*     ff_w3   = (const float*)d_in[11];
  const float*     ff_sc   = (const float*)d_in[12];
  float*           out     = (float*)d_out;               // [1000, 512]

  char* base = (char*)d_ws;
  const size_t szNode = (size_t)NNODES * DIMC * sizeof(float);   // 51.2 MB
  const size_t szY    = (size_t)NGRAPHS * EDIM * sizeof(float);  // 2.05 MB
  float* H     = (float*)(base);                 // node features
  float* AGGH2 = (float*)(base + 1 * szNode);    // AGG, then reused as H2
  float* T     = (float*)(base + 2 * szNode);    // MLP hidden
  float* Y     = (float*)(base + 3 * szNode);    // pooled [1000,512]
  float* t1    = (float*)(base + 3 * szNode + 1 * szY);
  float* t2    = (float*)(base + 3 * szNode + 2 * szY);
  float* t3    = (float*)(base + 3 * szNode + 3 * szY);
  float* stats = (float*)(base + 3 * szNode + 4 * szY);  // 256 f32
  float* ss    = stats + 256;                            // 256 f32

  const dim3 blk(256);
  const dim3 gNode((NNODES + 127) / 128, DIMC / 128);    // (782, 1)
  const dim3 gFF((NGRAPHS + 127) / 128, EDIM / 128);     // (8, 4)

  // zero pooled output accumulator
  zero_f32<<<(NGRAPHS * EDIM + 255) / 256, 256, 0, stream>>>(Y, NGRAPHS * EDIM);

  // pre: H = x @ pre_w + pre_b
  gemm_wmma_bf16<<<gNode, blk, 0, stream>>>(x, nullptr, pre_w, pre_b, nullptr, H,
                                            NNODES, DIMC, DIMC, 0);

  for (int l = 0; l < 4; ++l) {
    zero_f32<<<(NNODES * DIMC + 255) / 256, 256, 0, stream>>>(AGGH2, NNODES * DIMC);
    zero_f32<<<1, 256, 0, stream>>>(stats, 256);
    // agg = scatter_add over edges
    scatter_add<<<(NEDGES + 1) / 2, 256, 0, stream>>>(H, ei, AGGH2, NEDGES);
    // T = lrelu((H + agg) @ W1)
    gemm_wmma_bf16<<<gNode, blk, 0, stream>>>(H, AGGH2, conv_w1 + (size_t)l * DIMC * DIMC,
                                              nullptr, nullptr, T, NNODES, DIMC, DIMC, 1);
    // H2 = T @ W2   (AGG buffer reused as H2)
    gemm_wmma_bf16<<<gNode, blk, 0, stream>>>(T, nullptr, conv_w2 + (size_t)l * DIMC * DIMC,
                                              nullptr, nullptr, AGGH2, NNODES, DIMC, DIMC, 0);
    // BatchNorm stats + apply; fused per-graph add-pool into Y
    col_stats<<<512, 256, 0, stream>>>(AGGH2, stats, NNODES);
    bn_finalize<<<1, 128, 0, stream>>>(stats, bn_g + l * DIMC, bn_b + l * DIMC, ss, NNODES);
    bn_apply_pool<<<(NNODES * DIMC) / 256, 256, 0, stream>>>(AGGH2, ss, bat, H, Y, l, NNODES);
  }

  // FF head: z = lrelu(lrelu(lrelu(Y@W1)@W2)@W3) + Y@Wsc
  gemm_wmma_bf16<<<gFF, blk, 0, stream>>>(Y,  nullptr, ff_w1, nullptr, nullptr, t1,
                                          NGRAPHS, EDIM, EDIM, 1);
  gemm_wmma_bf16<<<gFF, blk, 0, stream>>>(t1, nullptr, ff_w2, nullptr, nullptr, t2,
                                          NGRAPHS, EDIM, EDIM, 1);
  gemm_wmma_bf16<<<gFF, blk, 0, stream>>>(t2, nullptr, ff_w3, nullptr, nullptr, t3,
                                          NGRAPHS, EDIM, EDIM, 1);
  gemm_wmma_bf16<<<gFF, blk, 0, stream>>>(Y,  nullptr, ff_sc, nullptr, t3, out,
                                          NGRAPHS, EDIM, EDIM, 0);
}
